// PositionAttentionModule_9320079032514
// MI455X (gfx1250) — matrix operand order repaired and checked by reading, not compiled, not run on hardware
//
#include <hip/hip_runtime.h>

// ---------------------------------------------------------------------------
// Position attention (DANet PAM) for MI455X / gfx1250, wave32 WMMA.
//   B=4, C=512, MID=64, H=W=64 -> N=4096
//   q = wq@x, k = wk@x, v = wv@x ; attn = softmax(q^T k) ; out = v attn^T
//   result = alpha*out + x
// Flash-attention formulation per batch:  O[q,d] = softmax(Q K^T)[q,:] . V^T[:,d]
// Q,K fp32 (v_wmma_f32_16x16x4_f32) for accurate logits,
// P.V^T in f16 (v_wmma_f32_16x16x32_f16) for throughput.
// Straight global->LDS staging uses GLOBAL_LOAD_ASYNC_TO_LDS_B128 (ASYNCcnt).
// ---------------------------------------------------------------------------

typedef __attribute__((ext_vector_type(16))) _Float16 v16h;
typedef __attribute__((ext_vector_type(8)))  _Float16 v8h;
typedef __attribute__((ext_vector_type(8)))  float    v8f;
typedef __attribute__((ext_vector_type(2)))  float    v2f;

union V8F  { v8f  v; float    f[8];  float4 q4[2]; };
union V16H { v16h v; v8h      h[2];  };

constexpr int Bsz  = 4;
constexpr int C    = 512;
constexpr int MIDD = 64;
constexpr int N    = 4096;   // 64*64

// ---- gfx1250 async global->LDS copy (16B per lane), tracked on ASYNCcnt ----
__device__ __forceinline__ void async_copy_b128(unsigned lds_off, const void* gaddr) {
  asm volatile("global_load_async_to_lds_b128 %0, %1, off"
               :: "v"(lds_off), "v"(gaddr) : "memory");
}
__device__ __forceinline__ void wait_async0() {
  asm volatile("s_wait_asynccnt 0x0" ::: "memory");
}
__device__ __forceinline__ unsigned lds_off_of(const void* p) {
  return (unsigned)(unsigned long long)p;   // low 32 bits of generic ptr = LDS offset
}

// ===========================================================================
// Kernel 1: Q / K GEMM, fp32 WMMA.  Out[n][m] = sum_c W[m][c]*x[b][c][n] + b[m]
// grid: (N/64, B, 2)  block: 256 (8 waves)
// ===========================================================================
__global__ __launch_bounds__(256) void qk_gemm_kernel(
    const float* __restrict__ x,
    const float* __restrict__ wq, const float* __restrict__ bq,
    const float* __restrict__ wk, const float* __restrict__ bk,
    float* __restrict__ Qb, float* __restrict__ Kb)
{
  const int nbase = blockIdx.x * 64;
  const int bb    = blockIdx.y;
  const float* W    = (blockIdx.z == 0) ? wq : wk;
  const float* bias = (blockIdx.z == 0) ? bq : bk;
  float* Out        = (blockIdx.z == 0) ? Qb : Kb;

  constexpr int ST = 36;                 // padded f32 row stride (16B aligned)
  __shared__ float sW[64 * ST];          // [m][c]
  __shared__ float sX[64 * ST];          // [n][c]  (x transposed)
  __shared__ float sBias[64];

  const int t = threadIdx.x;
  const int w = t >> 5, lane = t & 31, lm = lane & 15, g = lane >> 4;
  // 16 output tiles (4 mt x 4 nt), 2 per wave
  const int mts[2] = { (w * 2) >> 2, (w * 2 + 1) >> 2 };
  const int nts[2] = { (w * 2) & 3,  (w * 2 + 1) & 3  };

  if (t < 64) sBias[t] = bias[t];

  V8F acc[2];
  #pragma unroll
  for (int i = 0; i < 2; ++i)
    #pragma unroll
    for (int r = 0; r < 8; ++r) acc[i].f[r] = 0.f;

  for (int cs = 0; cs < 16; ++cs) {      // K = 512 in chunks of 32
    const int cbase = cs * 32;
    __syncthreads();
    {   // stage W chunk [64 m][32 c] via async copy (straight copy, f32)
      const int row = t >> 2, col = (t & 3) * 8;
      const float* src = W + (size_t)row * C + cbase + col;
      const unsigned dst = lds_off_of(&sW[row * ST + col]);
      async_copy_b128(dst,      src);
      async_copy_b128(dst + 16, src + 4);
    }
    {   // stage x chunk transposed -> sX[n][c] (needs VGPR transpose)
      const int crow = t >> 3, ncol = (t & 7) * 8;
      const float* src = x + ((size_t)(bb * C + cbase + crow)) * N + nbase + ncol;
      float4 a = *(const float4*)src;
      float4 b = *(const float4*)(src + 4);
      float tmp[8] = {a.x, a.y, a.z, a.w, b.x, b.y, b.z, b.w};
      #pragma unroll
      for (int i = 0; i < 8; ++i) sX[(ncol + i) * ST + crow] = tmp[i];
    }
    wait_async0();
    __syncthreads();
    #pragma unroll
    for (int ks = 0; ks < 8; ++ks) {     // 8 x (16x16x4) per 32-chunk
      const int cc = ks * 4 + 2 * g;
      #pragma unroll
      for (int i = 0; i < 2; ++i) {
        v2f a = *(const v2f*)&sW[(mts[i] * 16 + lm) * ST + cc];
        v2f b = *(const v2f*)&sX[(nts[i] * 16 + lm) * ST + cc];
        acc[i].v = __builtin_amdgcn_wmma_f32_16x16x4_f32(
            false, a, false, b, (short)0, acc[i].v, false, false);
      }
    }
  }
  // D: lane = N(n), vgpr r -> M(m) = r + 8g ; store rows Out[n][m] (contiguous m)
  #pragma unroll
  for (int i = 0; i < 2; ++i) {
    const int n  = nbase + nts[i] * 16 + lm;
    const int m0 = mts[i] * 16 + 8 * g;
    float* dst = Out + ((size_t)(bb * N + n)) * MIDD + m0;
    V8F r;
    #pragma unroll
    for (int rr = 0; rr < 8; ++rr) r.f[rr] = acc[i].f[rr] + sBias[m0 + rr];
    *(float4*)dst       = r.q4[0];
    *(float4*)(dst + 4) = r.q4[1];
  }
}

// ===========================================================================
// Kernel 2: V GEMM, f16 WMMA.  V[b][d][n] = f16( sum_c wv[d][c]*x[b][c][n] + bv[d] )
// grid: (N/64, C/64, B)  block: 256
// A = x^T (M=n,K=c), B = wv^T (K=c,N=d)  -> contiguous b128 stores along n
// (staging converts f32->f16, so no async path here)
// ===========================================================================
__global__ __launch_bounds__(256) void v_gemm_kernel(
    const float* __restrict__ x, const float* __restrict__ wv,
    const float* __restrict__ bv, _Float16* __restrict__ Vb)
{
  const int nbase = blockIdx.x * 64;
  const int dbase = blockIdx.y * 64;
  const int bb    = blockIdx.z;

  constexpr int ST = 40;                 // padded f16 row stride (16B aligned)
  __shared__ _Float16 sX[64 * ST];       // [n][c]
  __shared__ _Float16 sW[64 * ST];       // [d][c]

  const int t = threadIdx.x;
  const int w = t >> 5, lane = t & 31, lm = lane & 15, g = lane >> 4;
  const int nts[2] = { (w * 2) >> 2, (w * 2 + 1) >> 2 };
  const int dts[2] = { (w * 2) & 3,  (w * 2 + 1) & 3  };

  V8F acc[2];
  #pragma unroll
  for (int i = 0; i < 2; ++i)
    #pragma unroll
    for (int r = 0; r < 8; ++r) acc[i].f[r] = 0.f;

  for (int cs = 0; cs < 16; ++cs) {
    const int cbase = cs * 32;
    __syncthreads();
    {   // wv chunk [64 d][32 c] -> f16
      const int row = t >> 2, col = (t & 3) * 8;
      const float* src = wv + (size_t)(dbase + row) * C + cbase + col;
      float4 a = *(const float4*)src;
      float4 b = *(const float4*)(src + 4);
      float tmp[8] = {a.x, a.y, a.z, a.w, b.x, b.y, b.z, b.w};
      #pragma unroll
      for (int i = 0; i < 8; ++i) sW[row * ST + col + i] = (_Float16)tmp[i];
    }
    {   // x chunk transposed -> sX[n][c], f16
      const int crow = t >> 3, ncol = (t & 7) * 8;
      const float* src = x + ((size_t)(bb * C + cbase + crow)) * N + nbase + ncol;
      float4 a = *(const float4*)src;
      float4 b = *(const float4*)(src + 4);
      float tmp[8] = {a.x, a.y, a.z, a.w, b.x, b.y, b.z, b.w};
      #pragma unroll
      for (int i = 0; i < 8; ++i) sX[(ncol + i) * ST + crow] = (_Float16)tmp[i];
    }
    __syncthreads();
    #pragma unroll
    for (int i = 0; i < 2; ++i) {
      const int arow = (nts[i] * 16 + lm) * ST;   // A: M = n
      const int brow = (dts[i] * 16 + lm) * ST;   // B: N = d
      V16H a, b;
      a.h[0] = *(const v8h*)&sX[arow + 8 * g];
      a.h[1] = *(const v8h*)&sX[arow + 16 + 8 * g];
      b.h[0] = *(const v8h*)&sW[brow + 8 * g];
      b.h[1] = *(const v8h*)&sW[brow + 16 + 8 * g];
      acc[i].v = __builtin_amdgcn_wmma_f32_16x16x32_f16(
          false, a.v, false, b.v, (short)0, acc[i].v, false, false);
    }
  }
  // D: lane = N(d), vgpr r -> M(n)=r+8g ; store V[d][n] rows, contiguous n
  #pragma unroll
  for (int i = 0; i < 2; ++i) {
    const int d  = dbase + dts[i] * 16 + lm;
    const float bias = bv[d];
    const int n0 = nbase + nts[i] * 16 + 8 * g;
    _Float16* dst = Vb + ((size_t)(bb * C + d)) * N + n0;
    v8h r;
    #pragma unroll
    for (int rr = 0; rr < 8; ++rr) r[rr] = (_Float16)(acc[i].f[rr] + bias);
    *(v8h*)dst = r;
  }
}

// ===========================================================================
// Kernel 3: flash attention + epilogue.  grid: (N/64, B)  block: 256
// Each wave owns a 64-wide d slice; WG processes 64 queries, loops 64-key blocks.
// Q tile staged once and K tile staged per block with async LDS copies.
// ===========================================================================
__global__ __launch_bounds__(256, 1) void flash_attn_kernel(
    const float* __restrict__ Q, const float* __restrict__ K,
    const _Float16* __restrict__ V, const float* __restrict__ x,
    const float* __restrict__ alpha, float* __restrict__ out)
{
  const int qbase = blockIdx.x * 64;
  const int bb    = blockIdx.y;

  constexpr int QST = 68;   // f32 padded stride (16B aligned)
  constexpr int SST = 66;   // f32, scalar access only
  constexpr int PST = 72;   // f16 padded stride (16B aligned)
  __shared__ float    sQ[64 * QST];
  __shared__ float    sK[64 * QST];
  __shared__ float    sS[64 * SST];
  __shared__ _Float16 sP[64 * PST];
  __shared__ float    sM[64], sL[64], sScale[64];
  __shared__ float    sRed[256];

  const int t = threadIdx.x;
  const int w = t >> 5, lane = t & 31, lm = lane & 15, g = lane >> 4;
  // this wave's two S tiles (16 tiles total = 4 qt x 4 kt); sqt[0]==sqt[1]
  const int sqt    = (w * 2) >> 2;
  const int skt[2] = { (w * 2) & 3, (w * 2 + 1) & 3 };

  // staging geometry: 4 threads per row, 16 floats each (4 x b128)
  const int srow = t >> 2, scol = (t & 3) * 16;
  {   // stage Q tile [64 q][64 m] fp32 via async copy
    const float* src = Q + ((size_t)bb * N + qbase + srow) * MIDD + scol;
    const unsigned dst = lds_off_of(&sQ[srow * QST + scol]);
    #pragma unroll
    for (int i = 0; i < 4; ++i) async_copy_b128(dst + 16 * i, src + 4 * i);
  }
  if (t < 64) { sM[t] = -1e30f; sL[t] = 0.f; }
  wait_async0();
  __syncthreads();

  // O accumulators: [4 qt][4 dt] of 16x16 f32
  V8F o[16];
  #pragma unroll
  for (int i = 0; i < 16; ++i)
    #pragma unroll
    for (int r = 0; r < 8; ++r) o[i].f[r] = 0.f;

  for (int kb = 0; kb < 64; ++kb) {
    const int keyBase = kb * 64;

    // ---- stage K tile [64 key][64 m] via async copy ----
    {
      const float* src = K + ((size_t)bb * N + keyBase + srow) * MIDD + scol;
      const unsigned dst = lds_off_of(&sK[srow * QST + scol]);
      #pragma unroll
      for (int i = 0; i < 4; ++i) async_copy_b128(dst + 16 * i, src + 4 * i);
    }
    wait_async0();
    __syncthreads();

    // ---- step 1: S = Q K^T (fp32 WMMA), write to sS ----
    V8F s[2];
    #pragma unroll
    for (int i = 0; i < 2; ++i)
      #pragma unroll
      for (int r = 0; r < 8; ++r) s[i].f[r] = 0.f;
    #pragma unroll
    for (int ks = 0; ks < 16; ++ks) {
      const int cc = ks * 4 + 2 * g;
      v2f a = *(const v2f*)&sQ[(sqt * 16 + lm) * QST + cc];
      #pragma unroll
      for (int i = 0; i < 2; ++i) {
        v2f b = *(const v2f*)&sK[(skt[i] * 16 + lm) * QST + cc];
        s[i].v = __builtin_amdgcn_wmma_f32_16x16x4_f32(
            false, a, false, b, (short)0, s[i].v, false, false);
      }
    }
    #pragma unroll
    for (int i = 0; i < 2; ++i)
      #pragma unroll
      for (int r = 0; r < 8; ++r)
        sS[(sqt * 16 + r + 8 * g) * SST + skt[i] * 16 + lm] = s[i].f[r];
    __syncthreads();

    // ---- step 2: online softmax (4 threads per query row) ----
    {
      const int row = t >> 2, seg = t & 3;
      float* srowp = &sS[row * SST + seg * 16];
      float mx = -1e30f;
      #pragma unroll
      for (int j = 0; j < 16; ++j) mx = fmaxf(mx, srowp[j]);
      sRed[row * 4 + seg] = mx;
      __syncthreads();
      if (seg == 0) {
        float bm = fmaxf(fmaxf(sRed[row * 4], sRed[row * 4 + 1]),
                         fmaxf(sRed[row * 4 + 2], sRed[row * 4 + 3]));
        float mprev = sM[row];
        float mnew  = fmaxf(mprev, bm);
        sScale[row] = __expf(mprev - mnew);
        sM[row] = mnew;
      }
      __syncthreads();
      const float mnew = sM[row];
      float sum = 0.f;
      _Float16* prow = &sP[row * PST + seg * 16];
      #pragma unroll
      for (int j = 0; j < 16; ++j) {
        float p = __expf(srowp[j] - mnew);
        prow[j] = (_Float16)p;
        sum += p;
      }
      sRed[row * 4 + seg] = sum;
      __syncthreads();
      if (seg == 0)
        sL[row] = sL[row] * sScale[row] +
                  (sRed[row * 4] + sRed[row * 4 + 1] +
                   sRed[row * 4 + 2] + sRed[row * 4 + 3]);
      __syncthreads();
    }

    // ---- step 3: rescale O by exp(m_old - m_new) per query row ----
    float sc[32];
    #pragma unroll
    for (int qt = 0; qt < 4; ++qt)
      #pragma unroll
      for (int r = 0; r < 8; ++r)
        sc[qt * 8 + r] = sScale[qt * 16 + r + 8 * g];
    #pragma unroll
    for (int qt = 0; qt < 4; ++qt)
      #pragma unroll
      for (int dt = 0; dt < 4; ++dt)
        #pragma unroll
        for (int r = 0; r < 8; ++r) o[qt * 4 + dt].f[r] *= sc[qt * 8 + r];

    // ---- step 4: O += P . V^T (f16 WMMA); V frags loaded once per kk ----
    #pragma unroll
    for (int kk = 0; kk < 64; kk += 32) {
      V16H bfr[4];
      #pragma unroll
      for (int dt = 0; dt < 4; ++dt) {
        const int d = w * 64 + dt * 16 + lm;
        const _Float16* vrow =
            V + ((size_t)(bb * C + d)) * N + keyBase + kk + 8 * g;
        bfr[dt].h[0] = *(const v8h*)vrow;
        bfr[dt].h[1] = *(const v8h*)(vrow + 16);
      }
      #pragma unroll
      for (int qt = 0; qt < 4; ++qt) {
        V16H a;
        const int arow = (qt * 16 + lm) * PST + kk;
        a.h[0] = *(const v8h*)&sP[arow + 8 * g];
        a.h[1] = *(const v8h*)&sP[arow + 16 + 8 * g];
        #pragma unroll
        for (int dt = 0; dt < 4; ++dt) {
          o[qt * 4 + dt].v = __builtin_amdgcn_wmma_f32_16x16x32_f16(
              false, a.v, false, bfr[dt].v, (short)0, o[qt * 4 + dt].v,
              false, false);
        }
      }
    }
    __syncthreads();
  }

  // ---- epilogue: out = alpha * O / l + x ----
  const float al = alpha[0];
  #pragma unroll
  for (int qt = 0; qt < 4; ++qt) {
    float inv[8];
    #pragma unroll
    for (int r = 0; r < 8; ++r) inv[r] = 1.f / sL[qt * 16 + r + 8 * g];
    #pragma unroll
    for (int dt = 0; dt < 4; ++dt) {
      const int d  = w * 64 + dt * 16 + lm;
      const int q0 = qbase + qt * 16 + 8 * g;
      const size_t base = ((size_t)(bb * C + d)) * N + q0;
      float4 x0 = *(const float4*)(x + base);
      float4 x1 = *(const float4*)(x + base + 4);
      const float xs[8] = {x0.x, x0.y, x0.z, x0.w, x1.x, x1.y, x1.z, x1.w};
      V8F res;
      #pragma unroll
      for (int r = 0; r < 8; ++r)
        res.f[r] = al * (o[qt * 4 + dt].f[r] * inv[r]) + xs[r];
      *(float4*)(out + base)     = res.q4[0];
      *(float4*)(out + base + 4) = res.q4[1];
    }
  }
}

// ===========================================================================
extern "C" void kernel_launch(void* const* d_in, const int* in_sizes, int n_in,
                              void* d_out, int out_size, void* d_ws, size_t ws_size,
                              hipStream_t stream) {
  const float* x     = (const float*)d_in[0];
  const float* wq    = (const float*)d_in[1];
  const float* bq    = (const float*)d_in[2];
  const float* wk    = (const float*)d_in[3];
  const float* bk    = (const float*)d_in[4];
  const float* wv    = (const float*)d_in[5];
  const float* bv    = (const float*)d_in[6];
  const float* alpha = (const float*)d_in[7];
  float* out = (float*)d_out;

  // workspace: Q f32 (4MB) | K f32 (4MB) | V f16 (16MB)
  char* ws = (char*)d_ws;
  const size_t QK_BYTES = (size_t)Bsz * N * MIDD * sizeof(float);
  float*    Qb = (float*)ws;
  float*    Kb = (float*)(ws + QK_BYTES);
  _Float16* Vb = (_Float16*)(ws + 2 * QK_BYTES);

  qk_gemm_kernel<<<dim3(N / 64, Bsz, 2), 256, 0, stream>>>(x, wq, bq, wk, bk, Qb, Kb);
  v_gemm_kernel<<<dim3(N / 64, C / 64, Bsz), 256, 0, stream>>>(x, wv, bv, Vb);
  flash_attn_kernel<<<dim3(N / 64, Bsz), 256, 0, stream>>>(Qb, Kb, Vb, x, alpha, out);
}
